// DGC_33844342292506
// MI455X (gfx1250) — compile-verified, matrix operand order checked by reading, not run on hardware
//
#include <hip/hip_runtime.h>
#include <math.h>

typedef __attribute__((ext_vector_type(16))) __bf16 v16bf;
typedef __attribute__((ext_vector_type(8)))  __bf16 v8bf;
typedef __attribute__((ext_vector_type(8)))  float  v8f;

// ---------------------------------------------------------------------------
// Fragment loader: 16 bf16 per lane as two 16-byte loads (global_load_b128).
// A-layout (16x32 bf16): lane&15 = row, lane>>4 selects K-half:
//   lanes 0-15 : K = kb+0..7   and kb+16..23
//   lanes 16-31: K = kb+8..15  and kb+24..31
// pointer passed in is already offset by (lane>>4)*8 elements.
// B-layout mirrors A with columns (Bt stored [N,K] row-major).
// ---------------------------------------------------------------------------
__device__ __forceinline__ v16bf load_frag(const __bf16* p) {
  v8bf lo = *(const v8bf*)(p);
  v8bf hi = *(const v8bf*)(p + 16);
  return __builtin_shufflevector(lo, hi, 0, 1, 2, 3, 4, 5, 6, 7,
                                 8, 9, 10, 11, 12, 13, 14, 15);
}

// ---------------------------------------------------------------------------
// bf16 WMMA GEMM: C[M,N] = A[M,K] * B[K,N], A bf16 row-major, Bt bf16 [N,K].
// Block = 256 threads = 8 wave32s; block tile 128x128; wave tile 32x64.
// ---------------------------------------------------------------------------
__global__ __launch_bounds__(256) void k_gemm_bf16(
    const __bf16* __restrict__ A, const __bf16* __restrict__ Bt,
    float* __restrict__ C, int M, int N, int K) {
  const int lane = threadIdx.x & 31;
  const int wave = threadIdx.x >> 5;
  const int l15  = lane & 15;
  const int hsel = lane >> 4;  // 0 or 1 -> K-half offset of 8
  const int m0 = blockIdx.x * 128 + (wave & 3) * 32;
  const int n0 = blockIdx.y * 128 + (wave >> 2) * 64;

  int ra0 = m0 + l15;      if (ra0 >= M) ra0 = M - 1;
  int ra1 = m0 + 16 + l15; if (ra1 >= M) ra1 = M - 1;
  const __bf16* pa0 = A + (size_t)ra0 * K + hsel * 8;
  const __bf16* pa1 = A + (size_t)ra1 * K + hsel * 8;
  const __bf16* pb0 = Bt + (size_t)(n0 +  0 + l15) * K + hsel * 8;
  const __bf16* pb1 = Bt + (size_t)(n0 + 16 + l15) * K + hsel * 8;
  const __bf16* pb2 = Bt + (size_t)(n0 + 32 + l15) * K + hsel * 8;
  const __bf16* pb3 = Bt + (size_t)(n0 + 48 + l15) * K + hsel * 8;

  v8f acc[2][4];
#pragma unroll
  for (int i = 0; i < 2; ++i)
#pragma unroll
    for (int j = 0; j < 4; ++j) acc[i][j] = {};

  for (int kb = 0; kb < K; kb += 32) {
    // stream hints for the activation (A) rows
    __builtin_prefetch(pa0 + kb + 64, 0, 3);
    __builtin_prefetch(pa1 + kb + 64, 0, 3);

    v16bf a0 = load_frag(pa0 + kb);
    v16bf a1 = load_frag(pa1 + kb);
    v16bf b0 = load_frag(pb0 + kb);
    v16bf b1 = load_frag(pb1 + kb);
    v16bf b2 = load_frag(pb2 + kb);
    v16bf b3 = load_frag(pb3 + kb);

    acc[0][0] = __builtin_amdgcn_wmma_f32_16x16x32_bf16(false, a0, false, b0, (short)0, acc[0][0], false, false);
    acc[0][1] = __builtin_amdgcn_wmma_f32_16x16x32_bf16(false, a0, false, b1, (short)0, acc[0][1], false, false);
    acc[0][2] = __builtin_amdgcn_wmma_f32_16x16x32_bf16(false, a0, false, b2, (short)0, acc[0][2], false, false);
    acc[0][3] = __builtin_amdgcn_wmma_f32_16x16x32_bf16(false, a0, false, b3, (short)0, acc[0][3], false, false);
    acc[1][0] = __builtin_amdgcn_wmma_f32_16x16x32_bf16(false, a1, false, b0, (short)0, acc[1][0], false, false);
    acc[1][1] = __builtin_amdgcn_wmma_f32_16x16x32_bf16(false, a1, false, b1, (short)0, acc[1][1], false, false);
    acc[1][2] = __builtin_amdgcn_wmma_f32_16x16x32_bf16(false, a1, false, b2, (short)0, acc[1][2], false, false);
    acc[1][3] = __builtin_amdgcn_wmma_f32_16x16x32_bf16(false, a1, false, b3, (short)0, acc[1][3], false, false);
  }

  // C/D layout: VGPR g, lane l -> row = g + 8*(l>>4), col = l&15
#pragma unroll
  for (int i = 0; i < 2; ++i) {
#pragma unroll
    for (int j = 0; j < 4; ++j) {
      const int col = n0 + j * 16 + l15;
#pragma unroll
      for (int g = 0; g < 8; ++g) {
        const int row = m0 + i * 16 + g + 8 * hsel;
        if (row < M) C[(size_t)row * N + col] = acc[i][j][g];
      }
    }
  }
}

// ---------------------------------------------------------------------------
// Degree / dinv
// ---------------------------------------------------------------------------
__global__ void k_ones(float* p, int n) {
  int i = blockIdx.x * blockDim.x + threadIdx.x;
  if (i < n) p[i] = 1.0f;
}
__global__ void k_deg(const int* __restrict__ dst, float* deg, int E) {
  int i = blockIdx.x * blockDim.x + threadIdx.x;
  if (i < E) atomicAdd(&deg[dst[i]], 1.0f);
}
__global__ void k_rsqrt(float* p, int n) {
  int i = blockIdx.x * blockDim.x + threadIdx.x;
  if (i < n) p[i] = rsqrtf(p[i]);
}
__global__ void k_zero1(float* p) { if (threadIdx.x == 0) *p = 0.0f; }

// ---------------------------------------------------------------------------
// Conversions
// ---------------------------------------------------------------------------
__global__ void k_f32_to_bf16(const float* __restrict__ in, __bf16* __restrict__ out, long n) {
  long i = (long)blockIdx.x * blockDim.x + threadIdx.x;
  long s = (long)gridDim.x * blockDim.x;
  for (; i < n; i += s) out[i] = (__bf16)in[i];
}
// Wt[n*K + k] = (bf16) W[k*N + n]
__global__ void k_w_to_bf16_t(const float* __restrict__ W, __bf16* __restrict__ Wt, int K, int N) {
  long i = (long)blockIdx.x * blockDim.x + threadIdx.x;
  long s = (long)gridDim.x * blockDim.x;
  long tot = (long)K * N;
  for (; i < tot; i += s) {
    int k = (int)(i / N), n = (int)(i % N);
    Wt[(size_t)n * K + k] = (__bf16)W[i];
  }
}

// ---------------------------------------------------------------------------
// Propagation
// ---------------------------------------------------------------------------
__global__ void k_self_loop(const float* __restrict__ h, const float* __restrict__ dinv,
                            float* __restrict__ agg, int M, int F) {
  long i = (long)blockIdx.x * blockDim.x + threadIdx.x;
  long s = (long)gridDim.x * blockDim.x;
  long tot = (long)M * F;
  for (; i < tot; i += s) {
    int m = (int)(i / F);
    float d = dinv[m];
    agg[i] = h[i] * d * d;
  }
}

// one wave per edge: gather h[src] (float4, coalesced), atomic scatter to agg[dst]
__global__ __launch_bounds__(256) void k_scatter(
    const int* __restrict__ src, const int* __restrict__ dst,
    const float* __restrict__ dinv, const float* __restrict__ h,
    float* __restrict__ agg, int E, int F) {
  int wid  = (int)((blockIdx.x * (long)blockDim.x + threadIdx.x) >> 5);
  int lane = threadIdx.x & 31;
  if (wid >= E) return;
  int s = src[wid], d = dst[wid];
  float coef = dinv[s] * dinv[d];
  const float4* hs = (const float4*)(h + (size_t)s * F);
  float* ad = agg + (size_t)d * F;
  int f4n = F >> 2;
  for (int f4 = lane; f4 < f4n; f4 += 32) {
    float4 v = hs[f4];
    int f = f4 * 4;
    atomicAdd(ad + f + 0, v.x * coef);
    atomicAdd(ad + f + 1, v.y * coef);
    atomicAdd(ad + f + 2, v.z * coef);
    atomicAdd(ad + f + 3, v.w * coef);
  }
}

// layers 1,2: z = relu(agg + b) -> bf16 (input of next GEMM)
__global__ void k_bias_relu_bf16(const float* __restrict__ agg, const float* __restrict__ b,
                                 __bf16* __restrict__ out, int M, int F) {
  long i = (long)blockIdx.x * blockDim.x + threadIdx.x;
  long s = (long)gridDim.x * blockDim.x;
  long tot = (long)M * F;
  for (; i < tot; i += s) {
    int n = (int)(i % F);
    float v = agg[i] + b[n];
    out[i] = (__bf16)(v > 0.0f ? v : 0.0f);
  }
}

// layer 3: z = relu(agg + b) -> f32, plus global sum S (block-reduced)
__global__ __launch_bounds__(256) void k_bias_relu_sum(
    const float* __restrict__ agg, const float* __restrict__ b,
    float* __restrict__ z, float* __restrict__ S, int M, int F) {
  long i = (long)blockIdx.x * blockDim.x + threadIdx.x;
  long st = (long)gridDim.x * blockDim.x;
  long tot = (long)M * F;
  float local = 0.0f;
  for (; i < tot; i += st) {
    int n = (int)(i % F);
    float v = agg[i] + b[n];
    v = v > 0.0f ? v : 0.0f;
    z[i] = v;
    local += v;
  }
  for (int o = 16; o > 0; o >>= 1) local += __shfl_xor(local, o, 32);
  __shared__ float red[8];
  int lane = threadIdx.x & 31, wave = threadIdx.x >> 5;
  if (lane == 0) red[wave] = local;
  __syncthreads();
  if (threadIdx.x == 0) {
    float t = 0.0f;
    for (int w = 0; w < 8; ++w) t += red[w];
    atomicAdd(S, t);
  }
}

// final: t = tanh(z/S)^2 per element; out = t / max(||t_row||2, eps)
// one wave per row of 128 floats (float4 per lane)
__global__ __launch_bounds__(256) void k_finalize(
    const float* __restrict__ z, const float* __restrict__ S,
    float* __restrict__ out, int M) {
  int row  = blockIdx.x * 8 + (threadIdx.x >> 5);
  int lane = threadIdx.x & 31;
  if (row >= M) return;
  float invS = 1.0f / *S;
  const float4* zp = (const float4*)(z + (size_t)row * 128);
  float4 v = zp[lane];
  float t0 = tanhf(v.x * invS); t0 *= t0;
  float t1 = tanhf(v.y * invS); t1 *= t1;
  float t2 = tanhf(v.z * invS); t2 *= t2;
  float t3 = tanhf(v.w * invS); t3 *= t3;
  float ss = t0 * t0 + t1 * t1 + t2 * t2 + t3 * t3;
  for (int o = 16; o > 0; o >>= 1) ss += __shfl_xor(ss, o, 32);
  float nrm = sqrtf(ss);
  nrm = nrm > 1e-12f ? nrm : 1e-12f;
  float inv = 1.0f / nrm;
  float4 r = make_float4(t0 * inv, t1 * inv, t2 * inv, t3 * inv);
  ((float4*)(out + (size_t)row * 128))[lane] = r;
}

// ---------------------------------------------------------------------------
extern "C" void kernel_launch(void* const* d_in, const int* in_sizes, int n_in,
                              void* d_out, int out_size, void* d_ws, size_t ws_size,
                              hipStream_t stream) {
  const float* x  = (const float*)d_in[0];
  const int*   ei = (const int*)d_in[1];
  const float* W1 = (const float*)d_in[2];
  const float* b1 = (const float*)d_in[3];
  const float* W2 = (const float*)d_in[4];
  const float* b2 = (const float*)d_in[5];
  const float* W3 = (const float*)d_in[6];
  const float* b3 = (const float*)d_in[7];

  const int IN_DIM = 512, D1 = 512, D2 = 256, D3 = 128;
  const int M = in_sizes[0] / IN_DIM;       // 50000
  const int E = in_sizes[1] / 2;            // 400000
  const int* src = ei;
  const int* dst = ei + E;
  float* out = (float*)d_out;

  // workspace layout (256B aligned regions)
  char* ws = (char*)d_ws;
  size_t o = 0;
  float*  dinv  = (float*)(ws + o);  o += ((size_t)M * 4 + 255) & ~(size_t)255;
  float*  Ssum  = (float*)(ws + o);  o += 256;
  __bf16* inbf  = (__bf16*)(ws + o); o += (size_t)M * 512 * 2;            // activations (bf16), reused
  __bf16* Wt1   = (__bf16*)(ws + o); o += (size_t)IN_DIM * D1 * 2;
  __bf16* Wt2   = (__bf16*)(ws + o); o += (size_t)D1 * D2 * 2;
  __bf16* Wt3   = (__bf16*)(ws + o); o += (size_t)D2 * D3 * 2;
  float*  h     = (float*)(ws + o);  o += (size_t)M * 512 * 4;            // GEMM out, reused
  float*  agg   = (float*)(ws + o);  o += (size_t)M * 512 * 4;            // scatter target, reused
  (void)ws_size; (void)n_in; (void)out_size;

  const int TB = 256;
  auto eb = [&](long n) { long g = (n + TB - 1) / TB; return (int)(g < 16384 ? g : 16384); };

  // ---- degree / dinv ----
  k_ones<<<(M + TB - 1) / TB, TB, 0, stream>>>(dinv, M);
  k_deg<<<(E + TB - 1) / TB, TB, 0, stream>>>(dst, dinv, E);
  k_rsqrt<<<(M + TB - 1) / TB, TB, 0, stream>>>(dinv, M);
  k_zero1<<<1, 64, 0, stream>>>(Ssum);

  // ---- weight conversion (bf16, transposed) ----
  k_w_to_bf16_t<<<eb((long)IN_DIM * D1), TB, 0, stream>>>(W1, Wt1, IN_DIM, D1);
  k_w_to_bf16_t<<<eb((long)D1 * D2), TB, 0, stream>>>(W2, Wt2, D1, D2);
  k_w_to_bf16_t<<<eb((long)D2 * D3), TB, 0, stream>>>(W3, Wt3, D2, D3);

  // ---- layer 1 ----
  k_f32_to_bf16<<<eb((long)M * IN_DIM), TB, 0, stream>>>(x, inbf, (long)M * IN_DIM);
  {
    dim3 g((M + 127) / 128, D1 / 128);
    k_gemm_bf16<<<g, TB, 0, stream>>>(inbf, Wt1, h, M, D1, IN_DIM);
  }
  k_self_loop<<<eb((long)M * D1), TB, 0, stream>>>(h, dinv, agg, M, D1);
  k_scatter<<<(E + 7) / 8, TB, 0, stream>>>(src, dst, dinv, h, agg, E, D1);
  k_bias_relu_bf16<<<eb((long)M * D1), TB, 0, stream>>>(agg, b1, inbf, M, D1);

  // ---- layer 2 ----
  {
    dim3 g((M + 127) / 128, D2 / 128);
    k_gemm_bf16<<<g, TB, 0, stream>>>(inbf, Wt2, h, M, D2, D1);
  }
  k_self_loop<<<eb((long)M * D2), TB, 0, stream>>>(h, dinv, agg, M, D2);
  k_scatter<<<(E + 7) / 8, TB, 0, stream>>>(src, dst, dinv, h, agg, E, D2);
  k_bias_relu_bf16<<<eb((long)M * D2), TB, 0, stream>>>(agg, b2, inbf, M, D2);

  // ---- layer 3 ----
  {
    dim3 g((M + 127) / 128, D3 / 128);
    k_gemm_bf16<<<g, TB, 0, stream>>>(inbf, Wt3, h, M, D3, D2);
  }
  k_self_loop<<<eb((long)M * D3), TB, 0, stream>>>(h, dinv, agg, M, D3);
  k_scatter<<<(E + 7) / 8, TB, 0, stream>>>(src, dst, dinv, h, agg, E, D3);
  // z written over h (h no longer needed); S accumulated
  k_bias_relu_sum<<<eb((long)M * D3), TB, 0, stream>>>(agg, b3, h, Ssum, M, D3);

  // ---- finalize: tanh(z/S)^2 then row-L2 normalize ----
  k_finalize<<<(M + 7) / 8, TB, 0, stream>>>(h, Ssum, out, M);
}
